// S4Model_32452772888964
// MI455X (gfx1250) — matrix operand order repaired and checked by reading, not compiled, not run on hardware
//
#include <hip/hip_runtime.h>
#include <hip/hip_bf16.h>
#include <cstdint>

// ---------------------------------------------------------------------------
// S4D block forward for MI455X (gfx1250, wave32, WMMA).
//   B=16, L=4096, IN_DIM=1024, H=512, NC=16, N_CLASS=10
// Pipeline:
//   k0: fc1_w -> bf16 transposed (w1t[n][k]); conv_w -> bf16 (cwb[o][h])
//   k1: h = relu(x @ fc1_w + b)          WMMA bf16, block tile 128x128
//   k2: LayerNorm over H, in place
//   k3: S4D diagonal recurrence (16 complex modes/ch) + D skip + exact GELU
//   k4: z = y @ conv_w^T + cb, fused GLU  (a * sigmoid(b))   WMMA bf16
//   k5: maxpool over L + fc2 head -> logits (16 x 10) f32
// Workspace layout (bytes): buf0[0,128Mi) buf1[128Mi,256Mi) w1t cwb (2Mi)
// Requires ws_size >= 270,532,608 bytes.
// ---------------------------------------------------------------------------

typedef unsigned short u16;
typedef unsigned int   u32;
typedef __attribute__((ext_vector_type(16))) __bf16 v16bf;
typedef __attribute__((ext_vector_type(8)))  float  v8f;

union FragAB { v16bf v; u32 u[8]; };

__device__ inline u16 f2bf_bits(float f) {
  union { float f; u32 u; } v; v.f = f;
  u32 r = v.u + 0x7FFFu + ((v.u >> 16) & 1u);   // round-to-nearest-even
  return (u16)(r >> 16);
}

// A fragment: 16x32 bf16 tile, LDS row-major stride 32 bf16 (16 u32).
// lane<16: M=lane, VGPR v -> K pairs {0..7,16..23}; lane>=16: M=lane-16, K {8..15,24..31}
__device__ inline void load_a_frag(const u32* lA, int rowbase, int lane, FragAB& f) {
  int base = (rowbase + (lane & 15)) * 16 + ((lane & 16) ? 4 : 0);
#pragma unroll
  for (int v = 0; v < 8; ++v) f.u[v] = lA[base + (v & 3) + ((v >= 4) ? 8 : 0)];
}

// B fragment: 32x16 bf16 tile, LDS stored transposed: col n contiguous over K
// (stride 32 bf16 = 16 u32).  lane<16: N=lane, K {0..7,16..23}; lane>=16: N=lane-16, K {8..15,24..31}
__device__ inline void load_b_frag(const u32* lB, int colbase, int lane, FragAB& f) {
  int base = (colbase + (lane & 15)) * 16 + ((lane & 16) ? 4 : 0);
#pragma unroll
  for (int v = 0; v < 8; ++v) f.u[v] = lB[base + (v & 3) + ((v >= 4) ? 8 : 0)];
}

// ---------------------------------------------------------------------------
// k0: weight conversion. w1t[n*1024+k] = bf16(fc1_w[k*512+n]); cwb = bf16(conv_w)
__global__ __launch_bounds__(256) void k0_cvt_weights(
    const float* __restrict__ fc1w, const float* __restrict__ convw,
    u16* __restrict__ w1t, u16* __restrict__ cwb) {
  int idx = blockIdx.x * 256 + threadIdx.x;
  if (idx < 512 * 1024) {
    int n = idx >> 10, k = idx & 1023;
    w1t[idx] = f2bf_bits(fc1w[(size_t)k * 512 + n]);
    cwb[idx] = f2bf_bits(convw[idx]);
  }
}

// ---------------------------------------------------------------------------
// k1: h = relu(x @ fc1_w + fc1_b).  Block tile 128(M) x 128(N), 8 waves,
// each wave 32x64 (2x4 WMMA 16x16x32 bf16 tiles), K=1024 in 32 steps.
// Next k-tile is prefetched (global_prefetch) while WMMAs run.
__global__ __launch_bounds__(256) void k1_gemm1_relu(
    const float* __restrict__ X, const u16* __restrict__ Wt,
    const float* __restrict__ bias, float* __restrict__ Hout) {
  __shared__ u32 lA[128 * 16];   // 128 x 32 bf16 row-major
  __shared__ u32 lB[128 * 16];   // 128 cols x 32 k bf16 (transposed)
  u16* lAs = (u16*)lA;
  u16* lBs = (u16*)lB;
  const int tid = threadIdx.x, lane = tid & 31, wave = tid >> 5;
  const int tok0 = blockIdx.x * 128;
  const int n0   = blockIdx.y * 128;
  const int wm = (wave >> 1) * 32;   // wave row offset   (4 waves in M)
  const int wn = (wave & 1) * 64;    // wave col offset   (2 waves in N)
  v8f acc[2][4];
#pragma unroll
  for (int mi = 0; mi < 2; ++mi)
#pragma unroll
    for (int j = 0; j < 4; ++j) { v8f z = {}; acc[mi][j] = z; }

  for (int k0 = 0; k0 < 1024; k0 += 32) {
    __syncthreads();
#pragma unroll
    for (int i = 0; i < 16; ++i) {          // 128*32 fp32 -> bf16
      int e = tid + i * 256;
      int r = e >> 5, c = e & 31;
      lAs[e] = f2bf_bits(X[(size_t)(tok0 + r) * 1024 + k0 + c]);
    }
#pragma unroll
    for (int i = 0; i < 16; ++i) {          // 128*32 bf16 copy
      int e = tid + i * 256;
      int n = e >> 5, c = e & 31;
      lBs[e] = Wt[(size_t)(n0 + n) * 1024 + k0 + c];
    }
    // warm L2/L0 for the next k-tile while this one computes
    if (k0 + 32 < 1024) {
      int r = tid & 127;
      if (tid < 128) __builtin_prefetch(&X [(size_t)(tok0 + r) * 1024 + k0 + 32], 0, 3);
      else           __builtin_prefetch(&Wt[(size_t)(n0   + r) * 1024 + k0 + 32], 0, 3);
    }
    __syncthreads();
    FragAB af[2], bf[4];
    load_a_frag(lA, wm,      lane, af[0]);
    load_a_frag(lA, wm + 16, lane, af[1]);
#pragma unroll
    for (int j = 0; j < 4; ++j) load_b_frag(lB, wn + j * 16, lane, bf[j]);
#pragma unroll
    for (int mi = 0; mi < 2; ++mi)
#pragma unroll
      for (int j = 0; j < 4; ++j)
        acc[mi][j] = __builtin_amdgcn_wmma_f32_16x16x32_bf16(
            false, af[mi].v, false, bf[j].v, (short)0, acc[mi][j], false, false);
  }
  // C/D layout: VGPR r -> lanes 0-15: (M=r, N=lane); lanes 16-31: (M=8+r, N=lane-16)
#pragma unroll
  for (int mi = 0; mi < 2; ++mi)
#pragma unroll
    for (int j = 0; j < 4; ++j)
#pragma unroll
      for (int r = 0; r < 8; ++r) {
        int row = tok0 + wm + mi * 16 + r + ((lane & 16) ? 8 : 0);
        int col = n0 + wn + j * 16 + (lane & 15);
        float v = acc[mi][j][r] + bias[col];
        Hout[(size_t)row * 512 + col] = v > 0.f ? v : 0.f;
      }
}

// ---------------------------------------------------------------------------
// k2: LayerNorm over H=512, in place, one block per token.
__global__ __launch_bounds__(256) void k2_layernorm(
    float* __restrict__ Hb, const float* __restrict__ g, const float* __restrict__ bt) {
  __shared__ float s1[256], s2[256];
  const int t = blockIdx.x, tid = threadIdx.x;
  const size_t base = (size_t)t * 512;
  float a = Hb[base + tid], b = Hb[base + 256 + tid];
  s1[tid] = a + b;
  s2[tid] = a * a + b * b;
  __syncthreads();
  for (int off = 128; off > 0; off >>= 1) {
    if (tid < off) { s1[tid] += s1[tid + off]; s2[tid] += s2[tid + off]; }
    __syncthreads();
  }
  float mean = s1[0] * (1.f / 512.f);
  float var  = s2[0] * (1.f / 512.f) - mean * mean;
  float rstd = rsqrtf(var + 1e-5f);
  Hb[base + tid]       = (a - mean) * rstd * g[tid]       + bt[tid];
  Hb[base + 256 + tid] = (b - mean) * rstd * g[256 + tid] + bt[256 + tid];
}

// ---------------------------------------------------------------------------
// k3: S4D diagonal recurrence.  One thread per (b,h); 16 complex modes in VGPRs.
// y[l] = gelu( 2*Re(sum_n Ck_n s_n[l]) + Dp*u[l] ),  s_n[l] = w_n s_n[l-1] + u[l]
__global__ __launch_bounds__(128) void k3_s4d_scan(
    const float* __restrict__ U, const float* __restrict__ log_dt,
    const float* __restrict__ Are_, const float* __restrict__ Aim_,
    const float* __restrict__ Cre_, const float* __restrict__ Cim_,
    const float* __restrict__ Dp, float* __restrict__ Yout) {
  const int b = blockIdx.x;
  const int h = blockIdx.y * 128 + threadIdx.x;
  const float dt = expf(log_dt[h]);
  float wr[16], wi[16], cr[16], ci[16], sr[16], si[16];
#pragma unroll
  for (int n = 0; n < 16; ++n) {
    float are = -expf(Are_[h * 16 + n]);
    float aim = Aim_[h * 16 + n];
    float er  = expf(dt * are);
    float cth = cosf(dt * aim), sth = sinf(dt * aim);
    wr[n] = er * cth;
    wi[n] = er * sth;
    float dr  = wr[n] - 1.f, di = wi[n];
    float inv = 1.f / (are * are + aim * aim);
    float tr  = (dr * are + di * aim) * inv;     // (w-1)/A
    float ti  = (di * are - dr * aim) * inv;
    float Cre = Cre_[h * 16 + n], Cim = Cim_[h * 16 + n];
    cr[n] = 2.f * (Cre * tr - Cim * ti);
    ci[n] = 2.f * (Cre * ti + Cim * tr);
    sr[n] = 0.f; si[n] = 0.f;
  }
  const float dp = Dp[h];
  const size_t base = (size_t)b * 4096 * 512 + h;
  for (int l = 0; l < 4096; ++l) {
    float u = U[base + (size_t)l * 512];
    float acc = 0.f;
#pragma unroll
    for (int n = 0; n < 16; ++n) {
      float nsr = wr[n] * sr[n] - wi[n] * si[n] + u;
      float nsi = wr[n] * si[n] + wi[n] * sr[n];
      sr[n] = nsr; si[n] = nsi;
      acc += cr[n] * nsr - ci[n] * nsi;
    }
    float yv = acc + dp * u;
    yv = 0.5f * yv * (1.f + erff(yv * 0.70710678118f));   // exact GELU
    Yout[base + (size_t)l * 512] = yv;
  }
}

// ---------------------------------------------------------------------------
// k4: z = y @ conv_w^T + cb, fused GLU.  Block computes both the "a" tile
// (cols c0..c0+63) and its paired gate tile (cols 512+c0..512+c0+63);
// the A tile is shared so 8 WMMAs are fed by 6 fragment loads per k-step.
__global__ __launch_bounds__(256) void k4_gemm2_glu(
    const float* __restrict__ Y, const u16* __restrict__ Wc,
    const float* __restrict__ cb, float* __restrict__ G) {
  __shared__ u32 lA[128 * 16];
  __shared__ u32 lBa[64 * 16];
  __shared__ u32 lBb[64 * 16];
  u16* lAs  = (u16*)lA;
  u16* lBas = (u16*)lBa;
  u16* lBbs = (u16*)lBb;
  const int tid = threadIdx.x, lane = tid & 31, wave = tid >> 5;
  const int tok0 = blockIdx.x * 128;
  const int c0   = blockIdx.y * 64;
  const int wm = (wave >> 1) * 32;
  const int wn = (wave & 1) * 32;
  v8f aA[2][2], aB[2][2];
#pragma unroll
  for (int i = 0; i < 2; ++i) {
    v8f z = {};
    aA[i][0] = z; aA[i][1] = z; aB[i][0] = z; aB[i][1] = z;
  }
  for (int k0 = 0; k0 < 512; k0 += 32) {
    __syncthreads();
#pragma unroll
    for (int i = 0; i < 16; ++i) {
      int e = tid + i * 256;
      int r = e >> 5, c = e & 31;
      lAs[e] = f2bf_bits(Y[(size_t)(tok0 + r) * 512 + k0 + c]);
    }
#pragma unroll
    for (int i = 0; i < 8; ++i) {
      int e = tid + i * 256;
      int n = e >> 5, c = e & 31;
      lBas[e] = Wc[(size_t)(c0 + n) * 512 + k0 + c];
      lBbs[e] = Wc[(size_t)(512 + c0 + n) * 512 + k0 + c];
    }
    if (k0 + 32 < 512) {
      if (tid < 128) {
        __builtin_prefetch(&Y[(size_t)(tok0 + tid) * 512 + k0 + 32], 0, 3);
      } else {
        int n = tid - 128;
        if (n < 64) __builtin_prefetch(&Wc[(size_t)(c0 + n) * 512 + k0 + 32], 0, 3);
        else        __builtin_prefetch(&Wc[(size_t)(512 + c0 + (n - 64)) * 512 + k0 + 32], 0, 3);
      }
    }
    __syncthreads();
    FragAB a0, a1, ba0, ba1, bb0, bb1;
    load_a_frag(lA, wm,      lane, a0);
    load_a_frag(lA, wm + 16, lane, a1);
    load_b_frag(lBa, wn,      lane, ba0);
    load_b_frag(lBa, wn + 16, lane, ba1);
    load_b_frag(lBb, wn,      lane, bb0);
    load_b_frag(lBb, wn + 16, lane, bb1);
    aA[0][0] = __builtin_amdgcn_wmma_f32_16x16x32_bf16(false, a0.v, false, ba0.v, (short)0, aA[0][0], false, false);
    aA[0][1] = __builtin_amdgcn_wmma_f32_16x16x32_bf16(false, a0.v, false, ba1.v, (short)0, aA[0][1], false, false);
    aA[1][0] = __builtin_amdgcn_wmma_f32_16x16x32_bf16(false, a1.v, false, ba0.v, (short)0, aA[1][0], false, false);
    aA[1][1] = __builtin_amdgcn_wmma_f32_16x16x32_bf16(false, a1.v, false, ba1.v, (short)0, aA[1][1], false, false);
    aB[0][0] = __builtin_amdgcn_wmma_f32_16x16x32_bf16(false, a0.v, false, bb0.v, (short)0, aB[0][0], false, false);
    aB[0][1] = __builtin_amdgcn_wmma_f32_16x16x32_bf16(false, a0.v, false, bb1.v, (short)0, aB[0][1], false, false);
    aB[1][0] = __builtin_amdgcn_wmma_f32_16x16x32_bf16(false, a1.v, false, bb0.v, (short)0, aB[1][0], false, false);
    aB[1][1] = __builtin_amdgcn_wmma_f32_16x16x32_bf16(false, a1.v, false, bb1.v, (short)0, aB[1][1], false, false);
  }
#pragma unroll
  for (int mi = 0; mi < 2; ++mi)
#pragma unroll
    for (int ni = 0; ni < 2; ++ni)
#pragma unroll
      for (int r = 0; r < 8; ++r) {
        int row = tok0 + wm + mi * 16 + r + ((lane & 16) ? 8 : 0);
        int col = c0 + wn + ni * 16 + (lane & 15);
        float za = aA[mi][ni][r] + cb[col];
        float zb = aB[mi][ni][r] + cb[512 + col];
        float gl = za * (1.f / (1.f + expf(-zb)));
        G[(size_t)row * 512 + col] = gl;
      }
}

// ---------------------------------------------------------------------------
// k5: max over L, then tiny fc2 head. One block per batch.
__global__ __launch_bounds__(512) void k5_pool_fc2(
    const float* __restrict__ G, const float* __restrict__ W2,
    const float* __restrict__ b2, float* __restrict__ out) {
  __shared__ float pool[512];
  const int b = blockIdx.x, h = threadIdx.x;
  const size_t base = (size_t)b * 4096 * 512 + h;
  float m = -3.4e38f;
  for (int l = 0; l < 4096; ++l) m = fmaxf(m, G[base + (size_t)l * 512]);
  pool[h] = m;
  __syncthreads();
  if (h < 10) {
    float s = b2[h];
    for (int k = 0; k < 512; ++k) s += pool[k] * W2[k * 10 + h];
    out[b * 10 + h] = s;
  }
}

// ---------------------------------------------------------------------------
extern "C" void kernel_launch(void* const* d_in, const int* in_sizes, int n_in,
                              void* d_out, int out_size, void* d_ws, size_t ws_size,
                              hipStream_t stream) {
  (void)in_sizes; (void)n_in; (void)out_size; (void)ws_size;
  const float* x      = (const float*)d_in[0];
  const float* fc1_w  = (const float*)d_in[1];
  const float* fc1_b  = (const float*)d_in[2];
  const float* ln_g   = (const float*)d_in[3];
  const float* ln_b   = (const float*)d_in[4];
  const float* log_dt = (const float*)d_in[5];
  const float* A_re   = (const float*)d_in[6];
  const float* A_im   = (const float*)d_in[7];
  const float* C_re   = (const float*)d_in[8];
  const float* C_im   = (const float*)d_in[9];
  const float* Dp     = (const float*)d_in[10];
  const float* conv_w = (const float*)d_in[11];
  const float* conv_b = (const float*)d_in[12];
  const float* fc2_w  = (const float*)d_in[13];
  const float* fc2_b  = (const float*)d_in[14];
  float* out = (float*)d_out;

  char* ws = (char*)d_ws;
  float* buf0 = (float*)(ws);                         // 65536*512 f32 = 128 MiB
  float* buf1 = (float*)(ws + 134217728ull);          // 128 MiB
  u16*   w1t  = (u16*)  (ws + 268435456ull);          // 512*1024 bf16
  u16*   cwb  = (u16*)  (ws + 269484032ull);          // 1024*512 bf16

  k0_cvt_weights<<<dim3((512 * 1024 + 255) / 256), dim3(256), 0, stream>>>(fc1_w, conv_w, w1t, cwb);
  k1_gemm1_relu <<<dim3(512, 4), dim3(256), 0, stream>>>(x, w1t, fc1_b, buf0);
  k2_layernorm  <<<dim3(65536),  dim3(256), 0, stream>>>(buf0, ln_g, ln_b);
  k3_s4d_scan   <<<dim3(16, 4),  dim3(128), 0, stream>>>(buf0, log_dt, A_re, A_im, C_re, C_im, Dp, buf1);
  k4_gemm2_glu  <<<dim3(512, 8), dim3(256), 0, stream>>>(buf1, cwb, conv_b, buf0);
  k5_pool_fc2   <<<dim3(16),     dim3(512), 0, stream>>>(buf0, fc2_w, fc2_b, out);
}
